// SimpleMambaModel_38044820308606
// MI455X (gfx1250) — matrix operand order repaired and checked by reading, not compiled, not run on hardware
//
#include <hip/hip_runtime.h>
#include <hip/hip_bf16.h>

// ---------------- problem constants (match reference) ----------------
#define N_LAYERS 4
#define D_MODEL  1024
#define D_INNER  2048
#define D_STATE  16
#define D_CONV   4
#define DT_RANK  64
#define BATCH    2
#define SEQLEN   512
#define M_TOK    (BATCH * SEQLEN)        // 1024 token rows
#define XDBL_W   (DT_RANK + 2 * D_STATE) // 96

typedef unsigned short u16;

// ---------------- WMMA types ----------------
typedef __attribute__((ext_vector_type(16))) __bf16 v16bf;
typedef __attribute__((ext_vector_type(8)))  float  v8f;

union FragQ {
  v16bf v;
  uint4 q[2];
};

__device__ __forceinline__ u16 f2bf(float f) {
  // round-to-nearest-even fp32 -> bf16 (only used in elementwise/epilogue code)
  unsigned int u = __float_as_uint(f);
  return (u16)((u + 0x7FFFu + ((u >> 16) & 1u)) >> 16);
}

// A fragment: 16x32 bf16 from row-major bf16 A[M,K] (lda elements).
// Lane l: half=l>>4, r=l&15. Elements 0..7 = K (k0+half*8 .. +7),
// elements 8..15 = K (k0+16+half*8 .. +7): two contiguous 16B runs.
__device__ __forceinline__ v16bf loadA_bf(const u16* __restrict__ A, int lda,
                                          int row0, int k0, int lane) {
  int half = lane >> 4;
  int r    = lane & 15;
  const u16* base = A + (size_t)(row0 + r) * lda + k0 + half * 8;
  FragQ f;
  f.q[0] = *(const uint4*)(base);
  f.q[1] = *(const uint4*)(base + 16);
  return f.v;
}

// B fragment: 32x16 bf16, B[k][n] = W[n][k], W row-major bf16 (N,K) (ldw elems).
// Lane l: n=l&15, half=l>>4. Elements 0..7: K = k0 + half*8 + e; 8..15: K = k0 + 16 + half*8 + (e-8).
__device__ __forceinline__ v16bf loadB_bf(const u16* __restrict__ W, int ldw,
                                          int col0, int k0, int lane) {
  int half = lane >> 4;
  int n    = lane & 15;
  const u16* base = W + (size_t)(col0 + n) * ldw + k0 + half * 8;
  FragQ f;
  f.q[0] = *(const uint4*)(base);
  f.q[1] = *(const uint4*)(base + 16);
  return f.v;
}

__device__ __forceinline__ float epilogue(float x, int col,
                                          const float* __restrict__ bias, int act) {
  if (bias) x += bias[col];
  if (act == 1) {  // softplus
    x = (x > 20.f) ? x : log1pf(__expf(x));
  }
  return x;
}

#define WMMA_BF16(a, b, c) \
  __builtin_amdgcn_wmma_f32_16x16x32_bf16(false, (a), false, (b), (short)0, (c), false, false)

// C[M,N] = act( A[M,K](bf16) * B[N,K](bf16)^T + bias ).
// Optional fp32 output Cf and/or bf16 output Cbf (same ldc).
// Wave tile 64x32 (4 A-frags x 2 B-frags -> 8 WMMA per 32-wide K-step);
// block = 4 waves -> 128x64 tile.  Manual 2x-unrolled double buffer
// (requires K % 64 == 0): two static fragment register sets alternate, so
// loads for step k+32 / k+64 are in flight while WMMAs consume step k —
// no register rotation copies, no full loadcnt drains.
// All wave tiles are fully in-range or skipped wave-uniformly (EXEC all-ones).
__global__ __launch_bounds__(128)
void gemm_wmma_bf16(const u16* __restrict__ A, int lda,
                    const u16* __restrict__ B, int ldb,
                    float* __restrict__ Cf, u16* __restrict__ Cbf, int ldc,
                    int M, int N, int K,
                    const float* __restrict__ bias, int act) {
  const int lane = threadIdx.x & 31;
  const int w    = threadIdx.x >> 5;                    // 0..3
  const int row0 = blockIdx.y * 128 + (w >> 1) * 64;    // 64-row wave tile
  const int col0 = blockIdx.x * 64 + (w & 1) * 32;      // 32-col wave tile
  if (row0 >= M || col0 >= N) return;                   // wave-uniform skip

  v8f acc[4][2] = {};

  // buffer 0: fragments @ k = 0
  v16bf b00 = loadB_bf(B, ldb, col0,      0, lane);
  v16bf b01 = loadB_bf(B, ldb, col0 + 16, 0, lane);
  v16bf a0[4];
#pragma unroll
  for (int i = 0; i < 4; ++i) a0[i] = loadA_bf(A, lda, row0 + 16 * i, 0, lane);

  v16bf b10, b11;
  v16bf a1[4];

  for (int k0 = 0; k0 < K; k0 += 64) {
    // load buffer 1 @ k0+32 (always in range since K % 64 == 0)
    b10 = loadB_bf(B, ldb, col0,      k0 + 32, lane);
    b11 = loadB_bf(B, ldb, col0 + 16, k0 + 32, lane);
#pragma unroll
    for (int i = 0; i < 4; ++i) a1[i] = loadA_bf(A, lda, row0 + 16 * i, k0 + 32, lane);

    // consume buffer 0
#pragma unroll
    for (int i = 0; i < 4; ++i) {
      acc[i][0] = WMMA_BF16(a0[i], b00, acc[i][0]);
      acc[i][1] = WMMA_BF16(a0[i], b01, acc[i][1]);
    }

    // load buffer 0 @ k0+64 (wave-uniform guard on last iteration)
    if (k0 + 64 < K) {
      b00 = loadB_bf(B, ldb, col0,      k0 + 64, lane);
      b01 = loadB_bf(B, ldb, col0 + 16, k0 + 64, lane);
#pragma unroll
      for (int i = 0; i < 4; ++i) a0[i] = loadA_bf(A, lda, row0 + 16 * i, k0 + 64, lane);
      __builtin_prefetch(B + (size_t)(col0 + (lane & 15)) * ldb + k0 + 96, 0, 3);
    }

    // consume buffer 1
#pragma unroll
    for (int i = 0; i < 4; ++i) {
      acc[i][0] = WMMA_BF16(a1[i], b10, acc[i][0]);
      acc[i][1] = WMMA_BF16(a1[i], b11, acc[i][1]);
    }
  }

  const int half = lane >> 4;
  const int n    = lane & 15;
#pragma unroll
  for (int i = 0; i < 4; ++i) {
#pragma unroll
    for (int j = 0; j < 2; ++j) {
#pragma unroll
      for (int v = 0; v < 8; ++v) {
        int m = row0 + 16 * i + v + half * 8;
        int c = col0 + 16 * j + n;
        float val = epilogue(acc[i][j][v], c, bias, act);
        if (Cf)  Cf[(size_t)m * ldc + c]  = val;
        if (Cbf) Cbf[(size_t)m * ldc + c] = f2bf(val);
      }
    }
  }
}

// Vectorized fp32 -> bf16 conversion (8 elements / thread, b128 in/out).
__global__ __launch_bounds__(256)
void cvt_f32_bf16_kernel(const float* __restrict__ in, u16* __restrict__ out, int n8) {
  int i = blockIdx.x * blockDim.x + threadIdx.x;
  if (i >= n8) return;
  const float4* p = (const float4*)in + (size_t)i * 2;
  float4 a = p[0], b = p[1];
  union { uint4 q; u16 s[8]; } o;
  o.s[0] = f2bf(a.x); o.s[1] = f2bf(a.y); o.s[2] = f2bf(a.z); o.s[3] = f2bf(a.w);
  o.s[4] = f2bf(b.x); o.s[5] = f2bf(b.y); o.s[6] = f2bf(b.z); o.s[7] = f2bf(b.w);
  *((uint4*)out + i) = o.q;
}

// u[b,l,c] = silu( sum_j xz[b, l-3+j, c] * cw[c,j] + cb[c] ).  fp32 + bf16 out.
__global__ __launch_bounds__(256)
void conv_silu_kernel(const float* __restrict__ xz,
                      const float* __restrict__ cw,
                      const float* __restrict__ cb,
                      float* __restrict__ u, u16* __restrict__ ubf) {
  int idx = blockIdx.x * blockDim.x + threadIdx.x;
  if (idx >= M_TOK * D_INNER) return;
  int c = idx % D_INNER;
  int l = (idx / D_INNER) % SEQLEN;
  int b = idx / (D_INNER * SEQLEN);
  float acc = cb[c];
#pragma unroll
  for (int j = 0; j < D_CONV; ++j) {
    int t = l + j - (D_CONV - 1);
    if (t >= 0)
      acc += xz[(size_t)(b * SEQLEN + t) * (2 * D_INNER) + c] * cw[c * D_CONV + j];
  }
  acc = acc / (1.f + __expf(-acc));  // silu
  u[idx]   = acc;
  ubf[idx] = f2bf(acc);
}

// Sequential selective scan, one thread per (batch, channel). bf16 output.
__global__ __launch_bounds__(256)
void scan_kernel(const float* __restrict__ delta,
                 const float* __restrict__ u,
                 const float* __restrict__ xdbl,
                 const float* __restrict__ xz,
                 const float* __restrict__ A_log,
                 const float* __restrict__ Dp,
                 u16* __restrict__ ybf) {
  int idx = blockIdx.x * blockDim.x + threadIdx.x;
  if (idx >= BATCH * D_INNER) return;
  int d = idx % D_INNER;
  int b = idx / D_INNER;

  float A[D_STATE];
#pragma unroll
  for (int n = 0; n < D_STATE; ++n) A[n] = -__expf(A_log[d * D_STATE + n]);
  float h[D_STATE];
#pragma unroll
  for (int n = 0; n < D_STATE; ++n) h[n] = 0.f;
  const float dval = Dp[d];

  for (int l = 0; l < SEQLEN; ++l) {
    int row   = b * SEQLEN + l;
    float del = delta[(size_t)row * D_INNER + d];
    float uu  = u[(size_t)row * D_INNER + d];
    float du  = del * uu;
    const float* Brow = xdbl + (size_t)row * XDBL_W + DT_RANK;
    const float* Crow = Brow + D_STATE;
    float yacc = 0.f;
#pragma unroll
    for (int n = 0; n < D_STATE; ++n) {
      float dA = __expf(del * A[n]);
      h[n] = dA * h[n] + du * Brow[n];
      yacc += h[n] * Crow[n];
    }
    float z = xz[(size_t)row * (2 * D_INNER) + D_INNER + d];
    yacc = (yacc + uu * dval) * (z / (1.f + __expf(-z)));
    ybf[(size_t)row * D_INNER + d] = f2bf(yacc);
  }
}

static inline void cvt(const float* in, u16* out, size_t n, hipStream_t s) {
  int n8 = (int)(n / 8);
  cvt_f32_bf16_kernel<<<(n8 + 255) / 256, 256, 0, s>>>(in, out, n8);
}

extern "C" void kernel_launch(void* const* d_in, const int* in_sizes, int n_in,
                              void* d_out, int out_size, void* d_ws, size_t ws_size,
                              hipStream_t stream) {
  (void)in_sizes; (void)n_in; (void)out_size; (void)ws_size;
  const float* x         = (const float*)d_in[0];
  const float* in_proj_w = (const float*)d_in[1];
  const float* conv_w    = (const float*)d_in[2];
  const float* conv_b    = (const float*)d_in[3];
  const float* x_proj_w  = (const float*)d_in[4];
  const float* dt_proj_w = (const float*)d_in[5];
  const float* dt_proj_b = (const float*)d_in[6];
  const float* A_log     = (const float*)d_in[7];
  const float* Dw        = (const float*)d_in[8];
  const float* out_proj_w= (const float*)d_in[9];

  // ---- workspace carve-up (16B-aligned sections; ~57 MB total) ----
  char* cur = (char*)d_ws;
  auto carveF = [&](size_t n) { float* p = (float*)cur; cur += n * sizeof(float); return p; };
  auto carveH = [&](size_t n) { u16*   p = (u16*)cur;   cur += n * sizeof(u16);   return p; };

  float* xz    = carveF((size_t)M_TOK * 2 * D_INNER);  // 16 MB
  float* ubuf  = carveF((size_t)M_TOK * D_INNER);      // 8 MB
  float* xdbl  = carveF((size_t)M_TOK * XDBL_W);       // 384 KB
  float* delta = carveF((size_t)M_TOK * D_INNER);      // 8 MB

  u16* x_bf     = carveH((size_t)M_TOK * D_MODEL);     // 2 MB
  u16* ubf      = carveH((size_t)M_TOK * D_INNER);     // 4 MB
  u16* xdbl_bf  = carveH((size_t)M_TOK * XDBL_W);      // 192 KB
  u16* ybf      = carveH((size_t)M_TOK * D_INNER);     // 4 MB
  u16* xnext_bf = carveH((size_t)M_TOK * D_MODEL);     // 2 MB
  u16* w_in     = carveH((size_t)2 * D_INNER * D_MODEL); // 8 MB
  u16* w_x      = carveH((size_t)XDBL_W * D_INNER);    // 384 KB
  u16* w_dt     = carveH((size_t)D_INNER * DT_RANK);   // 256 KB
  u16* w_out    = carveH((size_t)D_MODEL * D_INNER);   // 4 MB

  dim3 blk(128);
  cvt(x, x_bf, (size_t)M_TOK * D_MODEL, stream);

  for (int i = 0; i < N_LAYERS; ++i) {
    // per-layer weight staging to bf16 (bandwidth-trivial elementwise kernels)
    cvt(in_proj_w + (size_t)i * 2 * D_INNER * D_MODEL, w_in, (size_t)2 * D_INNER * D_MODEL, stream);
    cvt(x_proj_w  + (size_t)i * XDBL_W * D_INNER,      w_x,  (size_t)XDBL_W * D_INNER,      stream);
    cvt(dt_proj_w + (size_t)i * D_INNER * DT_RANK,     w_dt, (size_t)D_INNER * DT_RANK,     stream);
    cvt(out_proj_w+ (size_t)i * D_MODEL * D_INNER,     w_out,(size_t)D_MODEL * D_INNER,     stream);

    const u16* xin_bf = (i == 0) ? x_bf : xnext_bf;

    // 1) xz = x @ Wi^T   (M=1024, N=4096, K=1024) -> fp32
    gemm_wmma_bf16<<<dim3(2 * D_INNER / 64, M_TOK / 128), blk, 0, stream>>>(
        xin_bf, D_MODEL, w_in, D_MODEL,
        xz, nullptr, 2 * D_INNER, M_TOK, 2 * D_INNER, D_MODEL, nullptr, 0);

    // 2) u = silu(causal_conv(xz[:, :D_INNER])) -> fp32 + bf16
    conv_silu_kernel<<<(M_TOK * D_INNER + 255) / 256, 256, 0, stream>>>(
        xz, conv_w + (size_t)i * D_INNER * D_CONV, conv_b + (size_t)i * D_INNER,
        ubuf, ubf);

    // 3) x_dbl = u @ Wx^T  (N=96) -> fp32 (scan B/C) + bf16 (dt GEMM A)
    gemm_wmma_bf16<<<dim3((XDBL_W + 63) / 64, M_TOK / 128), blk, 0, stream>>>(
        ubf, D_INNER, w_x, D_INNER,
        xdbl, xdbl_bf, XDBL_W, M_TOK, XDBL_W, D_INNER, nullptr, 0);

    // 4) delta = softplus(dt @ Wdt^T + bdt)  (A strided in x_dbl, K=64) -> fp32
    gemm_wmma_bf16<<<dim3(D_INNER / 64, M_TOK / 128), blk, 0, stream>>>(
        xdbl_bf, XDBL_W, w_dt, DT_RANK,
        delta, nullptr, D_INNER, M_TOK, D_INNER, DT_RANK,
        dt_proj_b + (size_t)i * D_INNER, 1);

    // 5) selective scan + D skip + silu(z) gate -> bf16 y
    scan_kernel<<<(BATCH * D_INNER + 255) / 256, 256, 0, stream>>>(
        delta, ubuf, xdbl, xz,
        A_log + (size_t)i * D_INNER * D_STATE, Dw + (size_t)i * D_INNER, ybf);

    // 6) out = y @ Wo^T  (M=1024, N=1024, K=2048)
    if (i == N_LAYERS - 1) {
      gemm_wmma_bf16<<<dim3(D_MODEL / 64, M_TOK / 128), blk, 0, stream>>>(
          ybf, D_INNER, w_out, D_INNER,
          (float*)d_out, nullptr, D_MODEL, M_TOK, D_MODEL, D_INNER, nullptr, 0);
    } else {
      gemm_wmma_bf16<<<dim3(D_MODEL / 64, M_TOK / 128), blk, 0, stream>>>(
          ybf, D_INNER, w_out, D_INNER,
          nullptr, xnext_bf, D_MODEL, M_TOK, D_MODEL, D_INNER, nullptr, 0);
    }
  }
}
